// LatentSDE_17463337026076
// MI455X (gfx1250) — matrix operand order repaired and checked by reading, not compiled, not run on hardware
//
#include <hip/hip_runtime.h>
#include <hip/hip_bf16.h>
#include <math.h>

typedef __bf16 bf16_t;
typedef __attribute__((ext_vector_type(16))) __bf16 v16bf;
typedef __attribute__((ext_vector_type(8)))  float  v8f;

#define T_  50
#define B_  1024
#define D_  32
#define L_  16
#define C_  64
#define H_  256
#define A_  8
#define NCHUNK 10
#define NSUB 16
#define NW  8                      // 8 wave32s per 256-thread block
#define HSUB (2.0f/(49.0f*4.0f))   // (TS[1]-TS[0])/4

#define WMMA_BF16(Aa,Bb,Cc) \
  __builtin_amdgcn_wmma_f32_16x16x32_bf16(false,(Aa),false,(Bb),(short)0,(Cc),false,false)

struct SDEParams {
  const float *xs, *actions, *rewards, *eps, *dW;
  const float *bih, *bhh, *elin_b, *qz0_b;
  const float *fb1, *fb2, *fb3, *hb1, *hb2, *hb3;
  const float *gw1, *gb1, *gw2, *gb2;
  const float *pb1, *pb2, *pb3, *act_b;
  const float *pzm, *pzl;
  const bf16_t *wihP, *whhP, *elinP, *qz0P;
  const bf16_t *fw1P, *fw2P, *fw3P, *hw1P, *hw2P, *hw3P;
  const bf16_t *pw1P, *pw2P, *pw3P, *actP;
  float *ctx, *rowLogp, *rowKl, *rowLq;
};

// ---------------------------------------------------------------------------
// Pack W (N x K, row major, fp32) into bf16 WMMA B-panels:
//   panel[(nt*ktiles+kt)*512 + lane*16 + i] = W[nt*16 + (lane&15)]
//                              [kt*32 + (i>>3)*16 + (lane>>4)*8 + (i&7)]
// (lanes 0-15: K 0..7,16..23, lanes 16-31: K 8..15,24..31, column = lane&15; zero pad OOB)
// ---------------------------------------------------------------------------
__global__ void pack_panels(const float* __restrict__ W, bf16_t* __restrict__ P,
                            int N, int K, int ntiles, int ktiles) {
  int tile = blockIdx.x;
  int nt = tile / ktiles, kt = tile % ktiles;
  int t = threadIdx.x;           // 512 threads
  int lane = t >> 4, i = t & 15;
  int n = nt * 16 + (lane & 15);
  int k = kt * 32 + ((i >> 3) << 4) + ((lane >> 4) << 3) + (i & 7);
  float v = (n < N && k < K) ? W[(size_t)n * K + k] : 0.f;
  P[((size_t)(nt * ktiles + kt)) * 512 + lane * 16 + i] = (bf16_t)v;
}

// Build one 16x32 bf16 A-fragment for this lane from fp32 X in LDS.
// xp = X + (l&15)*ldx + kt*32 + ((l>>4)*8): lane's data = two contiguous
// 8-float runs at +0 and +16 -> four ds_load_b128 instead of 16 scalar loads.
__device__ __forceinline__ v16bf packA(const float* __restrict__ xp) {
  float4 a0 = *(const float4*)(xp + 0);
  float4 a1 = *(const float4*)(xp + 4);
  float4 b0 = *(const float4*)(xp + 16);
  float4 b1 = *(const float4*)(xp + 20);
  v16bf r;
  r[0]  = (bf16_t)a0.x; r[1]  = (bf16_t)a0.y; r[2]  = (bf16_t)a0.z; r[3]  = (bf16_t)a0.w;
  r[4]  = (bf16_t)a1.x; r[5]  = (bf16_t)a1.y; r[6]  = (bf16_t)a1.z; r[7]  = (bf16_t)a1.w;
  r[8]  = (bf16_t)b0.x; r[9]  = (bf16_t)b0.y; r[10] = (bf16_t)b0.z; r[11] = (bf16_t)b0.w;
  r[12] = (bf16_t)b1.x; r[13] = (bf16_t)b1.y; r[14] = (bf16_t)b1.z; r[15] = (bf16_t)b1.w;
  return r;
}

// ---------------------------------------------------------------------------
// One 16-row GEMM tile stage: Y(16 x 16*ntiles) = act(X(16 x 32*KT) @ W^T + b)
// X fp32 in LDS (ldx multiple of 4, 16B-aligned base); B-panels bf16 from
// global (L2-resident). fp32 accumulation. Waves split N tiles; caller
// __syncthreads() between stages.
// ---------------------------------------------------------------------------
template<int KT, int ACT>
__device__ __forceinline__ void gemm16(const float* __restrict__ X, int ldx,
                                       const bf16_t* __restrict__ P, int ntiles,
                                       const float* __restrict__ bias, int biasN,
                                       float* __restrict__ Y, int ldy) {
  const int l  = threadIdx.x & 31;
  const int am = l & 15;
  const int kb = (l >> 4) << 3;
  const int m8 = (l >> 4) << 3;
  const int wave = threadIdx.x >> 5;
  v16bf a[KT];
#pragma unroll
  for (int kt = 0; kt < KT; ++kt)
    a[kt] = packA(X + am * ldx + kt * 32 + kb);
  for (int nt = wave; nt < ntiles; nt += NW) {
    v8f acc = {0.f,0.f,0.f,0.f,0.f,0.f,0.f,0.f};
    const bf16_t* bp = P + ((size_t)nt * KT) * 512 + l * 16;
#pragma unroll
    for (int kt = 0; kt < KT; ++kt) {
      v16bf b = *(const v16bf*)(bp + kt * 512);
      acc = WMMA_BF16(a[kt], b, acc);
    }
    int col = nt * 16 + am;
    float bv = (col < biasN) ? bias[col] : 0.f;
#pragma unroll
    for (int r = 0; r < 8; ++r) {
      float v = acc[r] + bv;
      if (ACT == 1) v = fmaxf(v, 0.f);
      if (ACT == 2) v = tanhf(v);
      Y[(m8 + r) * ldy + col] = v;
    }
  }
}

// evals(z, ctx_t): f = mlp3([z,ctx]) ; hdrift = mlp3(z) ; g = g_apply(z)
// l = 0.5*sum(((f-hdrift)/g)^2). Outputs fout/gout (16x16) + lout[16] in LDS.
__device__ void eval_fgl(const SDEParams& p, int tileBase,
                         float* bufA, float* bufB, float* xf, float* xh,
                         float* hdtmp,
                         const float* zin, int t,
                         float* fout, float* gout, float* lout) {
  const int tid = threadIdx.x;
  const float* ctx_t = p.ctx + ((size_t)t * B_ + tileBase) * C_;
  for (int e = tid; e < 16 * C_; e += 256) {
    int m = e >> 6, c = e & 63;
    xf[m * 96 + 16 + c] = ctx_t[m * C_ + c];
  }
  { int m = tid >> 4, l = tid & 15;
    float zv = zin[m * 16 + l];
    xf[m * 96 + l] = zv;
    xh[m * 32 + l] = zv; }
  __syncthreads();
  gemm16<3, 1>(xf,   96, p.fw1P, 16, p.fb1, H_, bufA, H_); __syncthreads();
  gemm16<8, 2>(bufA, H_, p.fw2P, 16, p.fb2, H_, bufB, H_); __syncthreads();
  gemm16<8, 0>(bufB, H_, p.fw3P,  1, p.fb3, L_, fout, 16); __syncthreads();
  gemm16<1, 1>(xh,   32, p.hw1P, 16, p.hb1, H_, bufA, H_); __syncthreads();
  gemm16<8, 2>(bufA, H_, p.hw2P, 16, p.hb2, H_, bufB, H_); __syncthreads();
  gemm16<8, 0>(bufB, H_, p.hw3P,  1, p.hb3, L_, hdtmp, 16);
  // g_apply: per (row,l) scalar reduction over H
  { int m = tid >> 4, l = tid & 15;
    float zv = zin[m * 16 + l];
    float acc = 0.f;
    for (int hh = 0; hh < H_; ++hh) {
      float t1 = zv * p.gw1[l * H_ + hh] + p.gb1[l * H_ + hh];
      if (t1 > 0.f) acc += t1 * p.gw2[l * H_ + hh];
    }
    gout[m * 16 + l] = acc + p.gb2[l]; }
  __syncthreads();
  if (tid < 16) {
    float s = 0.f;
    for (int l = 0; l < 16; ++l) {
      float u = (fout[tid * 16 + l] - hdtmp[tid * 16 + l]) / gout[tid * 16 + l];
      s += u * u;
    }
    lout[tid] = 0.5f * s;
  }
  __syncthreads();
}

// projection MLP on one z_out row-tile + log-likelihood accumulation vs target
__device__ void pred_logp(const SDEParams& p, int tileBase,
                          float* bufA, float* bufB, float* xproj,
                          const float* zsrc, int tg, float* rlogp) {
  const int tid = threadIdx.x;
  { int m = tid >> 4, l = tid & 15; xproj[m * 32 + l] = zsrc[m * 16 + l]; }
  __syncthreads();
  gemm16<1, 1>(xproj, 32, p.pw1P, 16, p.pb1, H_, bufA, H_); __syncthreads();
  gemm16<8, 2>(bufA,  H_, p.pw2P, 16, p.pb2, H_, bufB, H_); __syncthreads();
  gemm16<8, 0>(bufB,  H_, p.pw3P,  3, p.pb3, 33, bufA, 48); __syncthreads();
  for (int e = tid; e < 16 * 33; e += 256) {
    int m = e / 33, c = e - m * 33;
    size_t row = (size_t)tg * B_ + tileBase + m;
    float targ = (c < 32) ? p.xs[row * D_ + c] : p.rewards[row];
    float d = (targ - bufA[m * 48 + c]) * 100.f;              // / SIGMA
    float lp = -0.5f * d * d + 4.605170186f - 0.918938533f;   // -ln(sigma)-0.5 ln(2pi)
    atomicAdd(&rlogp[m], lp);
  }
  __syncthreads();
}

// ---------------------------------------------------------------------------
// Persistent kernel: one block = 16 batch rows, runs GRU scan, qz0/KL, and all
// 10 SDE chunks (row-local recurrences -> no inter-block sync needed).
// ---------------------------------------------------------------------------
__global__ __launch_bounds__(256) void latent_sde_main(SDEParams p) {
  __shared__ float sm[14960];
  const int tid = threadIdx.x;
  const int tileBase = blockIdx.x * 16;

  float* bufA  = sm;            // 4096 (GRU h double buffer / MLP scratch)
  float* bufB  = sm + 4096;     // 4096
  float* xf    = sm + 8192;     // 1536 (SDE f input 16x96; GRU: ctxTmp+xbuf)
  float* ctxTmp= xf;            // 1024 (GRU alias)
  float* xbuf  = xf + 1024;     // 512  (GRU alias)
  float* xh    = sm + 9728;     // 512
  float* xact  = sm + 10240;    // 1024
  float* xproj = sm + 11264;    // 512
  float* ctx0  = sm + 11776;    // 1024
  float* zb    = sm + 12800;    // 256
  float* zhb   = sm + 13056;    // 256
  float* zh1b  = sm + 13312;    // 256
  float* fhb   = sm + 13568;    // 256
  float* ghb   = sm + 13824;    // 256
  float* f1b   = sm + 14080;    // 256
  float* g1b   = sm + 14336;    // 256
  float* hdb   = sm + 14592;    // 256
  float* lq    = sm + 14848;    // 16
  float* lqh   = sm + 14864;    // 16
  float* lh    = sm + 14880;    // 16
  float* l1    = sm + 14896;    // 16
  float* rlogp = sm + 14912;    // 16
  float* rkl   = sm + 14928;    // 16
  float* rlq   = sm + 14944;    // 16

  for (int e = tid; e < 4096; e += 256) bufA[e] = 0.f;        // h0 = 0
  if (tid < 16) { rlogp[tid] = 0.f; rkl[tid] = 0.f; rlq[tid] = 0.f; }
  __syncthreads();

  float* hc = bufA; float* hn = bufB;

  // ---------------- GRU over reversed time, fused enc_lin -> ctx -----------
  for (int s = 0; s < T_; ++s) {
    const int tx = T_ - 1 - s;   // xs index and (flipped-back) ctx index
    for (int e = tid; e < 16 * D_; e += 256) {
      int m = e >> 5, k = e & 31;
      xbuf[m * D_ + k] = p.xs[((size_t)tx * B_ + tileBase + m) * D_ + k];
    }
    __syncthreads();
    {
      const int l = tid & 31, am = l & 15, kb8 = (l >> 4) << 3;
      const int m8 = (l >> 4) << 3;
      const int wave = tid >> 5;
      v16bf ah[8];
#pragma unroll
      for (int kt = 0; kt < 8; ++kt)
        ah[kt] = packA(hc + am * H_ + kt * 32 + kb8);
      v16bf ax = packA(xbuf + am * D_ + kb8);
      for (int jt = wave; jt < 16; jt += NW) {
        v8f aR = {0.f,0.f,0.f,0.f,0.f,0.f,0.f,0.f};
        v8f aZ = aR, aIN = aR, aHN = aR;
#pragma unroll
        for (int kt = 0; kt < 8; ++kt) {
          aR  = WMMA_BF16(ah[kt], *(const v16bf*)(p.whhP + ((size_t)( jt      * 8 + kt)) * 512 + l * 16), aR);
          aZ  = WMMA_BF16(ah[kt], *(const v16bf*)(p.whhP + ((size_t)((16 + jt) * 8 + kt)) * 512 + l * 16), aZ);
          aHN = WMMA_BF16(ah[kt], *(const v16bf*)(p.whhP + ((size_t)((32 + jt) * 8 + kt)) * 512 + l * 16), aHN);
        }
        aR  = WMMA_BF16(ax, *(const v16bf*)(p.wihP + (size_t)( jt)      * 512 + l * 16), aR);
        aZ  = WMMA_BF16(ax, *(const v16bf*)(p.wihP + (size_t)(16 + jt) * 512 + l * 16), aZ);
        aIN = WMMA_BF16(ax, *(const v16bf*)(p.wihP + (size_t)(32 + jt) * 512 + l * 16), aIN);
        int col = jt * 16 + am;
        float bR = p.bih[col]          + p.bhh[col];
        float bZ = p.bih[H_ + col]     + p.bhh[H_ + col];
        float bI = p.bih[2 * H_ + col];
        float bH = p.bhh[2 * H_ + col];
#pragma unroll
        for (int r = 0; r < 8; ++r) {
          int row = m8 + r;
          float rg = 1.f / (1.f + __expf(-(aR[r] + bR)));
          float zg = 1.f / (1.f + __expf(-(aZ[r] + bZ)));
          float ng = tanhf(aIN[r] + bI + rg * (aHN[r] + bH));
          float ho = hc[row * H_ + col];
          hn[row * H_ + col] = (1.f - zg) * ng + zg * ho;
        }
      }
    }
    __syncthreads();
    gemm16<8, 0>(hn, H_, p.elinP, 4, p.elin_b, C_, ctxTmp, C_);
    __syncthreads();
    for (int e = tid; e < 16 * C_; e += 256) {
      int m = e >> 6, c = e & 63;
      float v = ctxTmp[m * C_ + c];
      p.ctx[((size_t)tx * B_ + tileBase + m) * C_ + c] = v;
      if (tx == 0) ctx0[m * C_ + c] = v;
    }
    __threadfence_block();
    __syncthreads();
    float* t0 = hc; hc = hn; hn = t0;
  }

  // zero K-pad columns of all SDE input staging buffers (NaN safety)
  for (int e = tid; e < 3584; e += 256) xf[e] = 0.f;   // xf+xh+xact+xproj contiguous
  __syncthreads();

  // ---------------- qz0, z init, KL -----------------------------------------
  gemm16<2, 0>(ctx0, C_, p.qz0P, 2, p.qz0_b, 32, bufA, 32);
  __syncthreads();
  { int m = tid >> 4, l = tid & 15;
    float qm = bufA[m * 32 + l], ql = bufA[m * 32 + 16 + l];
    zb[m * 16 + l] = qm + __expf(ql) * p.eps[(size_t)(tileBase + m) * L_ + l];
    float pzl = p.pzl[l];
    float dq = ql - pzl;
    float dm = (qm - p.pzm[l]) / __expf(pzl);
    float kl = 0.5f * (__expf(2.f * dq) + dm * dm - 1.f) - dq;
    atomicAdd(&rkl[m], kl); }
  __syncthreads();

  // ---------------- SDE chunks ---------------------------------------------
  for (int c = 0; c < NCHUNK; ++c) {
    const int i0 = 5 * c;
    const int jx = (i0 == 0) ? 0 : i0 - 1;
    for (int e = tid; e < 16 * 56; e += 256) {
      int m = e / 56, cc = e - m * 56;
      float v;
      if (cc < 16)      v = zb[m * 16 + cc];
      else if (cc < 24) v = p.actions[((size_t)jx * B_ + tileBase + m) * A_ + (cc - 16)];
      else              v = p.xs[((size_t)jx * B_ + tileBase + m) * D_ + (cc - 24)];
      xact[m * 64 + cc] = v;
    }
    __syncthreads();
    gemm16<2, 0>(xact, 64, p.actP, 1, p.act_b, L_, zh1b, 16);  // z0c (temp)
    __syncthreads();
    { int m = tid >> 4, l = tid & 15;
      float v = zh1b[m * 16 + l];
      zb[m * 16 + l] = v; zhb[m * 16 + l] = v; }
    if (tid < 16) { lq[tid] = 0.f; lqh[tid] = 0.f; }
    __syncthreads();

    pred_logp(p, tileBase, bufA, bufB, xproj, zb, 5 * c, rlogp);
    eval_fgl(p, tileBase, bufA, bufB, xf, xh, hdb, zb, i0 + 1, fhb, ghb, lh);

    for (int k = 1; k <= NSUB; ++k) {
      float dWv;
      { int m = tid >> 4, l = tid & 15;
        int e = m * 16 + l;
        dWv = p.dW[(((size_t)c * NSUB + (k - 1)) * B_ + tileBase + m) * L_ + l];
        zh1b[e] = 2.f * zb[e] - zhb[e] + fhb[e] * HSUB + ghb[e] * dWv; }
      __syncthreads();
      int tk = i0 + 1 + (k >> 2); if (tk > T_ - 1) tk = T_ - 1;
      eval_fgl(p, tileBase, bufA, bufB, xf, xh, hdb, zh1b, tk, f1b, g1b, l1);
      { int m = tid >> 4, l = tid & 15;
        int e = m * 16 + l;
        float z1 = zb[e] + 0.5f * (fhb[e] + f1b[e]) * HSUB + 0.5f * (ghb[e] + g1b[e]) * dWv;
        zb[e] = z1; zhb[e] = zh1b[e]; fhb[e] = f1b[e]; ghb[e] = g1b[e]; }
      if (tid < 16) {
        float lqo = lq[tid];
        float lq1  = lqo + 0.5f * (lh[tid] + l1[tid]) * HSUB;
        float lqh1 = 2.f * lqo - lqh[tid] + lh[tid] * HSUB;
        lq[tid] = lq1; lqh[tid] = lqh1; lh[tid] = l1[tid];
      }
      __syncthreads();
      if ((k & 3) == 0)
        pred_logp(p, tileBase, bufA, bufB, xproj, zb, 5 * c + (k >> 2), rlogp);
    }
    if (tid < 16) rlq[tid] += lq[tid];   // telescoped log_ratio sum
    __syncthreads();
  }

  if (tid < 16) {
    int row = tileBase + tid;
    p.rowLogp[row] = rlogp[tid];
    p.rowKl[row]   = rkl[tid];
    p.rowLq[row]   = rlq[tid];
  }
}

__global__ void finalize_losses(const float* __restrict__ rl, const float* __restrict__ rk,
                                const float* __restrict__ rq, float* __restrict__ out, int B) {
  __shared__ float s[3];
  if (threadIdx.x < 3) s[threadIdx.x] = 0.f;
  __syncthreads();
  float a = 0.f, b = 0.f, c = 0.f;
  for (int r = threadIdx.x; r < B; r += blockDim.x) { a += rl[r]; b += rk[r]; c += rq[r]; }
  atomicAdd(&s[0], a); atomicAdd(&s[1], b); atomicAdd(&s[2], c);
  __syncthreads();
  if (threadIdx.x == 0) {
    out[0] = s[0] / (float)B;                    // log_pxs
    out[1] = s[1] / (float)B + s[2] / (float)B;  // logqp0 + logqp_path
  }
}

// ---------------------------------------------------------------------------
extern "C" void kernel_launch(void* const* d_in, const int* in_sizes, int n_in,
                              void* d_out, int out_size, void* d_ws, size_t ws_size,
                              hipStream_t stream) {
  (void)in_sizes; (void)n_in; (void)out_size; (void)ws_size;
  const float* xs      = (const float*)d_in[0];
  const float* actions = (const float*)d_in[1];
  const float* rewards = (const float*)d_in[2];
  const float* eps     = (const float*)d_in[3];
  const float* dW      = (const float*)d_in[4];
  const float* enc_wih = (const float*)d_in[5];
  const float* enc_whh = (const float*)d_in[6];
  const float* bih     = (const float*)d_in[7];
  const float* bhh     = (const float*)d_in[8];
  const float* elin_w  = (const float*)d_in[9];
  const float* elin_b  = (const float*)d_in[10];
  const float* qz0_w   = (const float*)d_in[11];
  const float* qz0_b   = (const float*)d_in[12];
  const float* fw1 = (const float*)d_in[13]; const float* fb1 = (const float*)d_in[14];
  const float* fw2 = (const float*)d_in[15]; const float* fb2 = (const float*)d_in[16];
  const float* fw3 = (const float*)d_in[17]; const float* fb3 = (const float*)d_in[18];
  const float* hw1 = (const float*)d_in[19]; const float* hb1 = (const float*)d_in[20];
  const float* hw2 = (const float*)d_in[21]; const float* hb2 = (const float*)d_in[22];
  const float* hw3 = (const float*)d_in[23]; const float* hb3 = (const float*)d_in[24];
  const float* gw1 = (const float*)d_in[25]; const float* gb1 = (const float*)d_in[26];
  const float* gw2 = (const float*)d_in[27]; const float* gb2 = (const float*)d_in[28];
  const float* pw1 = (const float*)d_in[29]; const float* pb1 = (const float*)d_in[30];
  const float* pw2 = (const float*)d_in[31]; const float* pb2 = (const float*)d_in[32];
  const float* pw3 = (const float*)d_in[33]; const float* pb3 = (const float*)d_in[34];
  const float* actw = (const float*)d_in[35]; const float* actb = (const float*)d_in[36];
  const float* pzm = (const float*)d_in[37]; const float* pzl = (const float*)d_in[38];

  char* ws = (char*)d_ws;
  size_t off = 0;
  auto take = [&](size_t bytes) -> char* {
    char* r = ws + off;
    off = (off + bytes + 1023) & ~(size_t)1023;
    return r;
  };
  auto panel = [&](int nt, int kt) -> bf16_t* {
    return (bf16_t*)take((size_t)nt * kt * 512 * sizeof(bf16_t));
  };
  bf16_t* wihP  = panel(48, 1);
  bf16_t* whhP  = panel(48, 8);
  bf16_t* elinP = panel(4, 8);
  bf16_t* qz0P  = panel(2, 2);
  bf16_t* fw1P  = panel(16, 3);
  bf16_t* fw2P  = panel(16, 8);
  bf16_t* fw3P  = panel(1, 8);
  bf16_t* hw1P  = panel(16, 1);
  bf16_t* hw2P  = panel(16, 8);
  bf16_t* hw3P  = panel(1, 8);
  bf16_t* pw1P  = panel(16, 1);
  bf16_t* pw2P  = panel(16, 8);
  bf16_t* pw3P  = panel(3, 8);
  bf16_t* actP  = panel(1, 2);
  float* ctx     = (float*)take((size_t)T_ * B_ * C_ * sizeof(float));
  float* rowLogp = (float*)take(B_ * sizeof(float));
  float* rowKl   = (float*)take(B_ * sizeof(float));
  float* rowLq   = (float*)take(B_ * sizeof(float));

  auto packW = [&](const float* W, bf16_t* P, int N, int K, int nt, int kt) {
    pack_panels<<<dim3(nt * kt), dim3(512), 0, stream>>>(W, P, N, K, nt, kt);
  };
  packW(enc_wih, wihP, 3 * H_, D_, 48, 1);
  packW(enc_whh, whhP, 3 * H_, H_, 48, 8);
  packW(elin_w, elinP, C_, H_, 4, 8);
  packW(qz0_w,  qz0P, 2 * L_, C_, 2, 2);
  packW(fw1, fw1P, H_, L_ + C_, 16, 3);
  packW(fw2, fw2P, H_, H_, 16, 8);
  packW(fw3, fw3P, L_, H_, 1, 8);
  packW(hw1, hw1P, H_, L_, 16, 1);
  packW(hw2, hw2P, H_, H_, 16, 8);
  packW(hw3, hw3P, L_, H_, 1, 8);
  packW(pw1, pw1P, H_, L_, 16, 1);
  packW(pw2, pw2P, H_, H_, 16, 8);
  packW(pw3, pw3P, 33, H_, 3, 8);
  packW(actw, actP, L_, L_ + A_ + D_, 1, 2);

  SDEParams p;
  p.xs = xs; p.actions = actions; p.rewards = rewards; p.eps = eps; p.dW = dW;
  p.bih = bih; p.bhh = bhh; p.elin_b = elin_b; p.qz0_b = qz0_b;
  p.fb1 = fb1; p.fb2 = fb2; p.fb3 = fb3; p.hb1 = hb1; p.hb2 = hb2; p.hb3 = hb3;
  p.gw1 = gw1; p.gb1 = gb1; p.gw2 = gw2; p.gb2 = gb2;
  p.pb1 = pb1; p.pb2 = pb2; p.pb3 = pb3; p.act_b = actb;
  p.pzm = pzm; p.pzl = pzl;
  p.wihP = wihP; p.whhP = whhP; p.elinP = elinP; p.qz0P = qz0P;
  p.fw1P = fw1P; p.fw2P = fw2P; p.fw3P = fw3P;
  p.hw1P = hw1P; p.hw2P = hw2P; p.hw3P = hw3P;
  p.pw1P = pw1P; p.pw2P = pw2P; p.pw3P = pw3P; p.actP = actP;
  p.ctx = ctx; p.rowLogp = rowLogp; p.rowKl = rowKl; p.rowLq = rowLq;

  latent_sde_main<<<dim3(B_ / 16), dim3(256), 0, stream>>>(p);
  finalize_losses<<<dim3(1), dim3(256), 0, stream>>>(rowLogp, rowKl, rowLq,
                                                     (float*)d_out, B_);
}
